// SimpleSSM_35167192219996
// MI455X (gfx1250) — matrix operand order repaired and checked by reading, not compiled, not run on hardware
//
#include <hip/hip_runtime.h>
#include <hip/hip_bf16.h>
#include <stdint.h>

// SimpleSSM on gfx1250:
//   Phase 1: bf16 WMMA projections; weight strip staged into LDS via the
//            Tensor Data Mover (tensor_load_to_lds + s_wait_tensorcnt),
//            4 M-tiles per wave reusing each B fragment 4x.
//   Phase 2: register-resident recurrent scan, lane n owns h[b,d,n].

typedef __attribute__((ext_vector_type(16))) __bf16 v16bf;
typedef __attribute__((ext_vector_type(8)))  __bf16 v8bf;
typedef __attribute__((ext_vector_type(8)))  float  v8f;
typedef __attribute__((ext_vector_type(4)))  unsigned int v4u;
typedef __attribute__((ext_vector_type(8)))  int    v8i;
typedef __attribute__((ext_vector_type(4)))  int    v4i;

#define BATCH  4
#define SEQ    2048
#define DIM    1024
#define DSTATE 32
#define MROWS  (BATCH * SEQ)   // 8192 rows fed to the projections
#define MT     4               // M-tiles per wave

#define HAVE_TDM __has_builtin(__builtin_amdgcn_tensor_load_to_lds)

// ---------------------------------------------------------------------------
// f32 -> bf16 conversion (one-time, feeds the WMMA units)
// ---------------------------------------------------------------------------
__global__ __launch_bounds__(256)
void cvt_f32_bf16_kernel(const float* __restrict__ src, __bf16* __restrict__ dst, int n) {
  int i = blockIdx.x * blockDim.x + threadIdx.x;
  if (i < n) dst[i] = (__bf16)src[i];
}

// ---------------------------------------------------------------------------
// TDM: DMA a 16-row x K-col bf16 weight strip (row-major, stride K) into LDS.
// D# per ISA 8.3/8.4: group0 = {count, lds_addr, global_addr, type=2},
// group1 = {data_size=2B, tensor_dim0=K, tensor_dim1=N, tile_dim0=K,
//           tile_dim1=16, tensor_dim0_stride=K}.  2D tile -> groups 2/3 zero.
// ---------------------------------------------------------------------------
#if HAVE_TDM
__device__ inline void tdm_load_weight_strip(const __bf16* gsrc, __bf16* lds_dst,
                                             int K, int N) {
  uint64_t ga = (uint64_t)(uintptr_t)gsrc;
  unsigned lds_off = (unsigned)(uintptr_t)lds_dst;
  v4u g0;
  g0[0] = 1u;                                                  // count=1 (valid D#)
  g0[1] = lds_off;                                             // lds_addr
  g0[2] = (unsigned)ga;                                        // global_addr[31:0]
  g0[3] = (unsigned)((ga >> 32) & 0x1FFFFFFu) | (2u << 30);    // addr[56:32] | type=2
  v8i g1;
  g1[0] = (int)(1u << 16);                                     // data_size=1 -> 2 bytes
  g1[1] = (int)((unsigned)(K & 0xffff) << 16);                 // tensor_dim0[15:0]
  g1[2] = (int)(((unsigned)K >> 16) |
                ((unsigned)(N & 0xffff) << 16));               // dim0[31:16] | dim1[15:0]
  g1[3] = (int)(((unsigned)N >> 16) |
                ((unsigned)(K & 0xffff) << 16));               // dim1[31:16] | tile_dim0=K
  g1[4] = 16;                                                  // tile_dim1=16 rows
  g1[5] = K;                                                   // tensor_dim0_stride[31:0]
  g1[6] = 0;
  g1[7] = 0;
  v4i z4 = {};
#if defined(__clang_major__) && (__clang_major__ >= 23)
  v8i z8 = {};
  __builtin_amdgcn_tensor_load_to_lds(g0, g1, z4, z4, z8, 0);
#else
  __builtin_amdgcn_tensor_load_to_lds(g0, g1, z4, z4, 0);
#endif
}
#endif

// ---------------------------------------------------------------------------
// out[M,N] = epilogue( Xbf[M,K] @ Wbf[N,K]^T + bias[N] )
// Block: 128 threads = 4 waves; block owns one 16-col N strip (weights in LDS)
// and 256 rows of M (4 waves x MT=4 tiles).  K-loop: 1 LDS B-fragment feeds
// MT wmma ops against MT global A-fragments.
// ---------------------------------------------------------------------------
__global__ __launch_bounds__(128)
void wmma_proj_kernel(const __bf16* __restrict__ Xbf, const __bf16* __restrict__ Wbf,
                      const float* __restrict__ bias, float* __restrict__ out,
                      int M, int N, int K, int do_softplus) {
  __shared__ __bf16 lds_w[16 * DIM];             // 32 KB weight strip (K <= 1024)

  const int lane    = threadIdx.x & 31;
  const int wave    = threadIdx.x >> 5;
  const int tile_n  = blockIdx.x;                // N / 16
  const int tile_m0 = blockIdx.y * 16 + wave * MT;
  const int hl      = lane >> 4;                 // wave half (K-offset select)
  const int l15     = lane & 15;

  // Stage the 16 x K weight strip into LDS once per block.
#if HAVE_TDM
  if (threadIdx.x < 32) {                        // one wave issues the DMA
    tdm_load_weight_strip(Wbf + (size_t)tile_n * 16 * K, lds_w, K, N);
    __builtin_amdgcn_s_wait_tensorcnt(0);
  }
#else
  for (int i = threadIdx.x * 8; i < 16 * K; i += blockDim.x * 8)
    *(v8bf*)&lds_w[i] = *(const v8bf*)&Wbf[(size_t)tile_n * 16 * K + i];
#endif
  __syncthreads();

  const int bcol = tile_n * 16 + l15;            // B/C/D column
  // B fragment source in LDS: lane column l15 -> row l15 of the strip,
  // lanes 0-15 K=kk..kk+15, lanes 16-31 K=kk+16..kk+31 (contiguous k order, same as A).
  const __bf16* lbrow = &lds_w[(size_t)l15 * K + hl * 16];

  v8f acc[MT];
#pragma unroll
  for (int t = 0; t < MT; ++t) acc[t] = (v8f){};

  for (int kk = 0; kk < K; kk += 32) {
    v16bf bfrag = *(const v16bf*)(lbrow + kk);   // ds_load_b128 x2, bank-clean

#pragma unroll
    for (int t = 0; t < MT; ++t) {
      // A fragment (16x32 bf16): lanes 0-15 hold K=kk..kk+15,
      // lanes 16-31 K=kk+16..kk+31.
      const int arow = (tile_m0 + t) * 16 + l15;
      const __bf16* ap = Xbf + (size_t)arow * K + kk + hl * 16;
      v8bf a_lo = *(const v8bf*)(ap);
      v8bf a_hi = *(const v8bf*)(ap + 8);
      v16bf afrag;
#pragma unroll
      for (int i = 0; i < 8; ++i) { afrag[i] = a_lo[i]; afrag[8 + i] = a_hi[i]; }

      acc[t] = __builtin_amdgcn_wmma_f32_16x16x32_bf16(false, afrag, false, bfrag,
                                                       (short)0, acc[t], false, false);
    }
  }

  // C/D layout: VGPR r -> M = r (lanes 0-15) or 8+r (lanes 16-31), N = lane&15.
  const float bv = bias[bcol];
#pragma unroll
  for (int t = 0; t < MT; ++t) {
#pragma unroll
    for (int r = 0; r < 8; ++r) {
      int m = (tile_m0 + t) * 16 + r + hl * 8;
      float v = acc[t][r] + bv;
      if (do_softplus) v = (v > 20.0f) ? v : __logf(1.0f + __expf(v));
      out[(size_t)m * N + bcol] = v;
    }
  }
}

// ---------------------------------------------------------------------------
// Recurrent scan. lane n (0..31) == state index n; each wave carries 4 adjacent
// d-channels so the b/c loads (shared across d within a batch) are amortized.
// h lives in registers for the whole sequence; y via 5-step shfl_xor butterfly.
// ---------------------------------------------------------------------------
__global__ __launch_bounds__(128)
void ssm_scan_kernel(const float* __restrict__ x,     const float* __restrict__ log_A,
                     const float* __restrict__ D_skip,
                     const float* __restrict__ dt,    const float* __restrict__ bproj,
                     const float* __restrict__ cproj, float* __restrict__ out) {
  const int lane = threadIdx.x & 31;               // state index n
  const int wave = threadIdx.x >> 5;
  const int bt   = blockIdx.y;                     // batch
  const int d0   = blockIdx.x * 16 + wave * 4;     // first of 4 d-channels

  float Arow[4], dsk[4], h[4];
#pragma unroll
  for (int j = 0; j < 4; ++j) {
    Arow[j] = -__expf(log_A[(size_t)(d0 + j) * DSTATE + lane]);  // A = -exp(log_A)
    dsk[j]  = D_skip[d0 + j];
    h[j]    = 0.0f;
  }

  const size_t base = (size_t)bt * SEQ;
  for (int s = 0; s < SEQ; ++s) {
    const size_t r = base + s;
    const float bn = bproj[r * DSTATE + lane];
    const float cn = cproj[r * DSTATE + lane];
    if (s + 8 < SEQ) {                             // global_prefetch_b8 ahead of use
      __builtin_prefetch(&bproj[(r + 8) * DSTATE + lane], 0, 1);
      __builtin_prefetch(&cproj[(r + 8) * DSTATE + lane], 0, 1);
      __builtin_prefetch(&dt[(r + 8) * DIM + d0], 0, 1);
      __builtin_prefetch(&x [(r + 8) * DIM + d0], 0, 1);
    }
#pragma unroll
    for (int j = 0; j < 4; ++j) {
      const float dts = dt[r * DIM + d0 + j];      // wave-uniform scalar loads
      const float xs  = x [r * DIM + d0 + j];
      h[j] = __expf(Arow[j] * dts) * h[j] + (dts * xs) * bn;  // A_bar*h + dt*x*b
      float p = h[j] * cn;
#pragma unroll
      for (int m = 16; m >= 1; m >>= 1) p += __shfl_xor(p, m, 32);
      if (lane == 0) out[r * DIM + d0 + j] = p + dsk[j] * xs;
    }
  }
}

// ---------------------------------------------------------------------------
extern "C" void kernel_launch(void* const* d_in, const int* in_sizes, int n_in,
                              void* d_out, int out_size, void* d_ws, size_t ws_size,
                              hipStream_t stream) {
  const float* x    = (const float*)d_in[0];   // [B,S,D]
  const float* logA = (const float*)d_in[1];   // [D,N]
  const float* B_w  = (const float*)d_in[2];   // [N,D]
  const float* B_b  = (const float*)d_in[3];   // [N]
  const float* C_w  = (const float*)d_in[4];   // [N,D]
  const float* C_b  = (const float*)d_in[5];   // [N]
  const float* Dsk  = (const float*)d_in[6];   // [D]
  const float* dt_w = (const float*)d_in[7];   // [D,D]
  const float* dt_b = (const float*)d_in[8];   // [D]
  float* out = (float*)d_out;                  // [B,S,D] f32

  // Workspace carve-out (~55 MB): bf16 operand copies + f32 dt/b/c intermediates.
  char*  ws  = (char*)d_ws;
  size_t off = 0;
  auto carve = [&](size_t bytes) -> void* {
    void* p = ws + off;
    off += (bytes + 255) & ~(size_t)255;
    return p;
  };
  __bf16* x_bf   = (__bf16*)carve((size_t)MROWS * DIM * sizeof(__bf16));
  __bf16* dtw_bf = (__bf16*)carve((size_t)DIM * DIM * sizeof(__bf16));
  __bf16* bw_bf  = (__bf16*)carve((size_t)DSTATE * DIM * sizeof(__bf16));
  __bf16* cw_bf  = (__bf16*)carve((size_t)DSTATE * DIM * sizeof(__bf16));
  float*  dt_f   = (float*)carve((size_t)MROWS * DIM * sizeof(float));
  float*  b_f    = (float*)carve((size_t)MROWS * DSTATE * sizeof(float));
  float*  c_f    = (float*)carve((size_t)MROWS * DSTATE * sizeof(float));

  // 1) bf16 conversions
  {
    int n = MROWS * DIM;
    cvt_f32_bf16_kernel<<<(n + 255) / 256, 256, 0, stream>>>(x, x_bf, n);
    n = DIM * DIM;
    cvt_f32_bf16_kernel<<<(n + 255) / 256, 256, 0, stream>>>(dt_w, dtw_bf, n);
    n = DSTATE * DIM;
    cvt_f32_bf16_kernel<<<(n + 255) / 256, 256, 0, stream>>>(B_w, bw_bf, n);
    cvt_f32_bf16_kernel<<<(n + 255) / 256, 256, 0, stream>>>(C_w, cw_bf, n);
  }

  // 2) WMMA projections: block = 4 waves x MT(4) tiles = 256 rows x 16 cols.
  {
    dim3 blk(128, 1, 1);
    dim3 grd_dt(DIM / 16, MROWS / 256, 1);     // 64 x 32 blocks
    wmma_proj_kernel<<<grd_dt, blk, 0, stream>>>(x_bf, dtw_bf, dt_b, dt_f,
                                                 MROWS, DIM, DIM, /*softplus=*/1);
    dim3 grd_bc(DSTATE / 16, MROWS / 256, 1);  // 2 x 32 blocks
    wmma_proj_kernel<<<grd_bc, blk, 0, stream>>>(x_bf, bw_bf, B_b, b_f,
                                                 MROWS, DSTATE, DIM, /*softplus=*/0);
    wmma_proj_kernel<<<grd_bc, blk, 0, stream>>>(x_bf, cw_bf, C_b, c_f,
                                                 MROWS, DSTATE, DIM, /*softplus=*/0);
  }

  // 3) Recurrent scan: 4 waves/block * 4 d each -> 16 d per block
  {
    dim3 blk(128, 1, 1);
    dim3 grd(DIM / 16, BATCH, 1);
    ssm_scan_kernel<<<grd, blk, 0, stream>>>(x, logA, Dsk, dt_f, b_f, c_f, out);
  }
}
